// CapNet_55259049230660
// MI455X (gfx1250) — matrix operand order repaired and checked by reading, not compiled, not run on hardware
//
#include <hip/hip_runtime.h>
#include <hip/hip_bf16.h>

// ---------- types for WMMA ----------
typedef __attribute__((ext_vector_type(16))) __bf16 v16bf;
typedef __attribute__((ext_vector_type(8)))  float  v8f;
typedef unsigned int v4u __attribute__((ext_vector_type(4)));
typedef int          v4i __attribute__((ext_vector_type(4)));
typedef int          v8i __attribute__((ext_vector_type(8)));

struct alignas(16) U128 { unsigned int x, y, z, w; };
union FragBF { U128 q[2]; v16bf v; };

__device__ __forceinline__ unsigned short f2bf(float f) {
    union { float f; unsigned int u; } x; x.f = f;
    unsigned int r = x.u + 0x7FFFu + ((x.u >> 16) & 1u);  // round-nearest-even
    return (unsigned short)(r >> 16);
}

// ---------- TDM: async 2D bf16 tile DMA global -> LDS (CDNA5 tensor_load_to_lds) ----------
// D# per CDNA5 ISA: group0 = {count, lds_addr, global_addr, type=2},
// group1 = {data_size=2B, tensor dims, tile dims, dim0 stride}; groups 2/3 zero (2D tile).
__device__ __forceinline__ void tdm_load_tile_bf16(
    const void* gsrc, unsigned ldsOff,
    unsigned tile0 /*elems*/, unsigned tile1 /*rows*/,
    unsigned tdim0, unsigned tdim1, unsigned long long stride0 /*elems*/)
{
    unsigned long long ga = (unsigned long long)(uintptr_t)gsrc;
    v4u g0;
    g0[0] = 1u;                                             // count=1, user descriptor
    g0[1] = ldsOff;                                         // lds_addr (bytes)
    g0[2] = (unsigned)ga;                                   // global_addr[31:0]
    g0[3] = (unsigned)((ga >> 32) & 0x01FFFFFFu) | (2u << 30); // addr[56:32] | type=2
    v8i g1;
    g1[0] = (int)(1u << 16);                                // data_size=1 (2 bytes)
    g1[1] = (int)(tdim0 << 16);                             // tensor_dim0[15:0] @ [63:48]
    g1[2] = (int)((tdim0 >> 16) | (tdim1 << 16));           // td0 hi | td1 lo
    g1[3] = (int)((tdim1 >> 16) | (tile0 << 16));           // td1 hi | tile_dim0
    g1[4] = (int)(tile1 & 0xFFFFu);                         // tile_dim1 | tile_dim2=0
    g1[5] = (int)(unsigned)(stride0 & 0xFFFFFFFFu);         // dim0_stride[31:0]
    g1[6] = (int)(unsigned)((stride0 >> 32) & 0xFFFFu);     // dim0_stride[47:32]
    g1[7] = 0;
    v4i gz = {0, 0, 0, 0};
#if defined(__clang_major__) && (__clang_major__ >= 23)
    v8i gz8 = {0, 0, 0, 0, 0, 0, 0, 0};
    __builtin_amdgcn_tensor_load_to_lds(g0, g1, gz, gz, gz8, 0);
#else
    __builtin_amdgcn_tensor_load_to_lds(g0, g1, gz, gz, 0);
#endif
}

// ---------- sizes ----------
#define BATCH 512
#define K1    256        // conv1 K padded (real 243), k = (kh*9+kw)*3+ic
#define K1R   243
#define N1    (512*400)
#define K2    20736      // k = (kh*9+kw)*256+ic
#define N2    (512*36)
#define NCAPS 1152

#define OCT1  32
#define PXT1  64
#define OCT2  64
#define PXT2  128

// ---------- weight prep ----------
__global__ __launch_bounds__(256) void prep_w1(const float* __restrict__ w,
                                               unsigned short* __restrict__ w1n) {
    int idx = blockIdx.x * 256 + threadIdx.x;          // 256*256
    int oc = idx >> 8, k = idx & 255;
    unsigned short v = 0;
    if (k < K1R) {
        int kpos = k / 3, ic = k - kpos * 3;
        v = f2bf(w[(oc * 3 + ic) * 81 + kpos]);
    }
    w1n[idx] = v;
}

__global__ __launch_bounds__(256) void prep_w2(const float* __restrict__ w,
                                               unsigned short* __restrict__ w2n) {
    int idx = blockIdx.x * 256 + threadIdx.x;          // 256*20736
    int oc = idx / K2, k = idx - oc * K2;
    int kpos = k >> 8, ic = k & 255;
    w2n[idx] = f2bf(w[(oc * 256 + ic) * 81 + kpos]);
}

// ---------- conv1: implicit GEMM (bf16 WMMA), fused ReLU, NHWC bf16 out ----------
__global__ __launch_bounds__(256) void conv1_wmma(
    const unsigned short* __restrict__ w1n,   // [256][256] bf16
    const float* __restrict__ x,              // [512][3][28][28]
    const float* __restrict__ bias,           // [256]
    unsigned short* __restrict__ h1n)         // [512][20][20][256] bf16 post-relu (NHWC)
{
    __shared__ unsigned short ldsA[OCT1][32];
    __shared__ unsigned short ldsB[PXT1][32];
    const int tid  = threadIdx.x;
    const int lane = tid & 31;
    const int wave = tid >> 5;
    const int ocBase = blockIdx.y * OCT1;
    const int nBase  = blockIdx.x * PXT1;
    const int ocSub  = ocBase + (wave & 1) * 16;
    const int pxSub  = nBase + (wave >> 1) * 16;

    int kkJ[8], cbJ[8], ldsOffJ[8];
#pragma unroll
    for (int j = 0; j < 8; ++j) {
        int i = tid + j * 256;
        int px = i >> 5, kk = i & 31;
        int n = nBase + px;
        int b = n / 400, p = n - b * 400;
        int oh = p / 20, ow = p - oh * 20;
        kkJ[j] = kk;
        cbJ[j] = b * 2352 + oh * 28 + ow;
        ldsOffJ[j] = px * 32 + kk;
    }

    v8f acc = {};
    for (int kb = 0; kb < K1; kb += 32) {
        __syncthreads();
        if (tid < 128) {
            int row = tid >> 2, c4 = tid & 3;
            *(uint4*)&ldsA[row][c4 * 8] =
                *(const uint4*)(w1n + (ocBase + row) * K1 + kb + c4 * 8);
        }
        float vals[8];
#pragma unroll
        for (int j = 0; j < 8; ++j) {
            int k = kb + kkJ[j];
            float f = 0.0f;
            if (k < K1R) {
                int kpos = k / 3, ic = k - kpos * 3;
                int kh = kpos / 9, kw = kpos - kh * 9;
                f = x[cbJ[j] + ic * 784 + kh * 28 + kw];
            }
            vals[j] = f;
        }
#pragma unroll
        for (int j = 0; j < 8; ++j)
            ((unsigned short*)&ldsB[0][0])[ldsOffJ[j]] = f2bf(vals[j]);
        if (kb + 32 < K1)
            __builtin_prefetch(w1n + (ocBase + lane) * K1 + kb + 32, 0, 1);
        __syncthreads();

        FragBF fa, fb;
        {
            int row = (wave & 1) * 16 + (lane & 15);
            int kh8 = (lane >> 4) * 8;
            fa.q[0] = *(const U128*)&ldsA[row][kh8];
            fa.q[1] = *(const U128*)&ldsA[row][kh8 + 16];
            int nrow = (wave >> 1) * 16 + (lane & 15);
            int kb16 = (lane >> 4) * 8;
            fb.q[0] = *(const U128*)&ldsB[nrow][kb16];
            fb.q[1] = *(const U128*)&ldsB[nrow][kb16 + 16];
        }
        acc = __builtin_amdgcn_wmma_f32_16x16x32_bf16(
            false, fa.v, false, fb.v, (short)0, acc, false, false);
    }

    int nOut = pxSub + (lane & 15);
    int oc0  = ocSub + 8 * (lane >> 4);
    unsigned int pk[4];
#pragma unroll
    for (int rr = 0; rr < 4; ++rr) {
        float v0 = acc[2 * rr]     + bias[oc0 + 2 * rr];
        float v1 = acc[2 * rr + 1] + bias[oc0 + 2 * rr + 1];
        v0 = v0 > 0.0f ? v0 : 0.0f;
        v1 = v1 > 0.0f ? v1 : 0.0f;
        pk[rr] = (unsigned int)f2bf(v0) | ((unsigned int)f2bf(v1) << 16);
    }
    *(uint4*)(h1n + (size_t)nOut * 256 + oc0) = make_uint4(pk[0], pk[1], pk[2], pk[3]);
}

// ---------- conv2: 64x128 tile, TDM-fed double-buffered A, reg double-buffered B ----------
__global__ __launch_bounds__(256) void conv2_wmma(
    const unsigned short* __restrict__ w2n,   // [256][20736] bf16
    const unsigned short* __restrict__ h1n,   // [512][20][20][256] bf16 NHWC
    const float* __restrict__ bias,           // [256]
    float* __restrict__ s2)                   // [512][9216] f32 (oc*36+p)
{
    __shared__ unsigned short ldsA2[2][OCT2][32];   // 8 KB, TDM double buffer
    __shared__ unsigned short ldsB[PXT2][32];       // 8 KB
    const int tid  = threadIdx.x;
    const int lane = tid & 31;
    const int wave = tid >> 5;
    const int ocBase = blockIdx.y * OCT2;
    const int nBase  = blockIdx.x * PXT2;
    const int ocw = wave & 1;
    const int pxw = wave >> 1;

    // B staging geometry: 128 px x 64B = 512 b128 chunks, two per thread
    size_t bBase[2];
    uint4* lB[2];
#pragma unroll
    for (int j = 0; j < 2; ++j) {
        int i = tid + j * 256;
        int px = i >> 2, c4 = i & 3;
        int n = nBase + px;
        int b = n / 36, p = n - b * 36;
        int oh = p / 6, ow = p - oh * 6;
        bBase[j] = (size_t)(b * 400 + oh * 40 + ow * 2) * 256 + c4 * 8;
        lB[j] = (uint4*)&ldsB[px][c4 * 8];
    }

    const unsigned ldsAOff[2] = {
        (unsigned)(uintptr_t)(void*)&ldsA2[0][0][0],
        (unsigned)(uintptr_t)(void*)&ldsA2[1][0][0]
    };
    const unsigned short* aTile = w2n + (size_t)ocBase * K2;

    v8f acc00 = {}, acc01 = {}, acc10 = {}, acc11 = {};

    // prologue: TDM A block 0, vector-load B block 0 (kh=kw=0, icBase=0)
    if (wave == 0)
        tdm_load_tile_bf16(aTile, ldsAOff[0], 32, OCT2, K2, 256, (unsigned long long)K2);
    uint4 rB0 = *(const uint4*)(h1n + bBase[0]);
    uint4 rB1 = *(const uint4*)(h1n + bBase[1]);
    *lB[0] = rB0; *lB[1] = rB1;

    for (int kb = 0; kb < K2; kb += 32) {
        const int buf = (kb >> 5) & 1;
        __builtin_amdgcn_s_wait_tensorcnt(0);   // producer drains DMA; others no-op
        __syncthreads();

        FragBF a0, a1, b0, b1;
        {
            int ra = ocw * 32 + (lane & 15);
            int kh8 = (lane >> 4) * 8;
            a0.q[0] = *(const U128*)&ldsA2[buf][ra][kh8];
            a0.q[1] = *(const U128*)&ldsA2[buf][ra][kh8 + 16];
            a1.q[0] = *(const U128*)&ldsA2[buf][ra + 16][kh8];
            a1.q[1] = *(const U128*)&ldsA2[buf][ra + 16][kh8 + 16];
            int rb = pxw * 32 + (lane & 15);
            int kb16 = (lane >> 4) * 8;
            b0.q[0] = *(const U128*)&ldsB[rb][kb16];
            b0.q[1] = *(const U128*)&ldsB[rb][kb16 + 16];
            b1.q[0] = *(const U128*)&ldsB[rb + 16][kb16];
            b1.q[1] = *(const U128*)&ldsB[rb + 16][kb16 + 16];
        }

        const bool more = (kb + 32) < K2;
        if (more) {
            int kn = kb + 32;
            if (wave == 0)
                tdm_load_tile_bf16(aTile + kn, ldsAOff[buf ^ 1],
                                   32, OCT2, K2, 256, (unsigned long long)K2);
            int kpos = kn >> 8, icB = kn & 255;
            int kh = kpos / 9, kw = kpos - kh * 9;
            size_t bOff = (size_t)((kh * 20 + kw) * 256 + icB);
            rB0 = *(const uint4*)(h1n + bBase[0] + bOff);
            rB1 = *(const uint4*)(h1n + bBase[1] + bOff);
        }

        acc00 = __builtin_amdgcn_wmma_f32_16x16x32_bf16(false, a0.v, false, b0.v, (short)0, acc00, false, false);
        acc01 = __builtin_amdgcn_wmma_f32_16x16x32_bf16(false, a0.v, false, b1.v, (short)0, acc01, false, false);
        acc10 = __builtin_amdgcn_wmma_f32_16x16x32_bf16(false, a1.v, false, b0.v, (short)0, acc10, false, false);
        acc11 = __builtin_amdgcn_wmma_f32_16x16x32_bf16(false, a1.v, false, b1.v, (short)0, acc11, false, false);
        __syncthreads();
        if (more) { *lB[0] = rB0; *lB[1] = rB1; }
    }

    auto storeTile = [&](v8f a, int iSub, int jSub) {
        int nOut = nBase + pxw * 32 + jSub * 16 + (lane & 15);
        int bb = nOut / 36, p = nOut - bb * 36;
        int oc0 = ocBase + ocw * 32 + iSub * 16 + 8 * (lane >> 4);
#pragma unroll
        for (int r = 0; r < 8; ++r)
            s2[bb * 9216 + (oc0 + r) * 36 + p] = a[r] + bias[oc0 + r];
    };
    storeTile(acc00, 0, 0);
    storeTile(acc01, 0, 1);
    storeTile(acc10, 1, 0);
    storeTile(acc11, 1, 1);
}

// ---------- primary capsule squash ----------
__global__ __launch_bounds__(256) void squash_primary(const float* __restrict__ s2,
                                                      float* __restrict__ u) {
    int idx = blockIdx.x * 256 + threadIdx.x;
    const float* src = s2 + idx * 8;
    float d[8], sn = 0.0f;
#pragma unroll
    for (int i = 0; i < 8; ++i) { d[i] = src[i]; sn = fmaf(d[i], d[i], sn); }
    float sc = sn / (1.0f + sn) / (sqrtf(sn) + 1e-6f);
    float* dst = u + idx * 8;
#pragma unroll
    for (int i = 0; i < 8; ++i) dst[i] = d[i] * sc;
}

__global__ __launch_bounds__(256) void zero_f32(float* __restrict__ p, int n) {
    int idx = blockIdx.x * 256 + threadIdx.x;
    if (idx < n) p[idx] = 0.0f;
}

// ---------- routing: s = sum_cap softmax(b)*u_hat ; v = squash(s) ----------
__global__ __launch_bounds__(256) void routing_sv(
    const float* __restrict__ u, const float* __restrict__ W,
    const float* __restrict__ blog, float* __restrict__ v)
{
    __shared__ float sAcc[10][16];
    const int b = blockIdx.x;
    const int tid = threadIdx.x;
    if (tid < 160) ((float*)sAcc)[tid] = 0.0f;
    __syncthreads();

    const int e = tid & 15;
    const int g = tid >> 4;
    float sPart[10];
#pragma unroll
    for (int o = 0; o < 10; ++o) sPart[o] = 0.0f;

    for (int cap = g; cap < NCAPS; cap += 16) {
        const float* bl = blog + (b * NCAPS + cap) * 10;
        float l[10], mx = bl[0];
#pragma unroll
        for (int o = 0; o < 10; ++o) { l[o] = bl[o]; mx = fmaxf(mx, l[o]); }
        float sum = 0.0f;
#pragma unroll
        for (int o = 0; o < 10; ++o) { l[o] = __expf(l[o] - mx); sum += l[o]; }
        float inv = 1.0f / sum;
        const float* uc = u + (b * NCAPS + cap) * 8;
        float ud[8];
#pragma unroll
        for (int d = 0; d < 8; ++d) ud[d] = uc[d];
        const float* Wc = W + cap * 1280 + e;
#pragma unroll
        for (int o = 0; o < 10; ++o) {
            float uh = 0.0f;
#pragma unroll
            for (int d = 0; d < 8; ++d) uh = fmaf(ud[d], Wc[(o * 8 + d) * 16], uh);
            sPart[o] = fmaf(l[o] * inv, uh, sPart[o]);
        }
    }
#pragma unroll
    for (int o = 0; o < 10; ++o) atomicAdd(&sAcc[o][e], sPart[o]);
    __syncthreads();

    if (tid < 160) {
        int o = tid / 16, ee = tid % 16;
        float sn = 0.0f;
#pragma unroll
        for (int q = 0; q < 16; ++q) { float t = sAcc[o][q]; sn = fmaf(t, t, sn); }
        float sc = sn / (1.0f + sn) / (sqrtf(sn) + 1e-6f);
        v[(b * 10 + o) * 16 + ee] = sAcc[o][ee] * sc;
    }
}

// ---------- routing: b += sum_e u_hat * v ----------
__global__ __launch_bounds__(256) void routing_blog(
    const float* __restrict__ u, const float* __restrict__ W,
    const float* __restrict__ v, float* __restrict__ blog)
{
    int idx = blockIdx.x * 256 + threadIdx.x;
    if (idx >= BATCH * NCAPS * 10) return;
    int o = idx % 10, cap = (idx / 10) % NCAPS, b = idx / (NCAPS * 10);
    const float* uc = u + (b * NCAPS + cap) * 8;
    const float* Wc = W + (cap * 10 + o) * 128;
    const float* vc = v + (b * 10 + o) * 16;
    float acc = 0.0f;
#pragma unroll
    for (int e = 0; e < 16; ++e) {
        float uh = 0.0f;
#pragma unroll
        for (int d = 0; d < 8; ++d) uh = fmaf(uc[d], Wc[d * 16 + e], uh);
        acc = fmaf(uh, vc[e], acc);
    }
    blog[idx] += acc;
}

// ---------- launch ----------
extern "C" void kernel_launch(void* const* d_in, const int* in_sizes, int n_in,
                              void* d_out, int out_size, void* d_ws, size_t ws_size,
                              hipStream_t stream) {
    const float* x   = (const float*)d_in[0];
    const float* w1  = (const float*)d_in[1];
    const float* b1  = (const float*)d_in[2];
    const float* w2  = (const float*)d_in[3];
    const float* b2  = (const float*)d_in[4];
    const float* W   = (const float*)d_in[5];
    float* out = (float*)d_out;

    char* ws = (char*)d_ws;
    size_t off = 0;
    auto alloc = [&](size_t bytes) {
        size_t o = off;
        off += (bytes + 255) & ~(size_t)255;
        return o;
    };
    unsigned short* w1n  = (unsigned short*)(ws + alloc((size_t)256 * K1 * 2));
    unsigned short* w2n  = (unsigned short*)(ws + alloc((size_t)256 * K2 * 2));
    unsigned short* h1n  = (unsigned short*)(ws + alloc((size_t)BATCH * 400 * 256 * 2));
    float*          s2   = (float*)(ws + alloc((size_t)BATCH * 9216 * 4));
    float*          u    = (float*)(ws + alloc((size_t)BATCH * NCAPS * 8 * 4));
    float*          blog = (float*)(ws + alloc((size_t)BATCH * NCAPS * 10 * 4));
    float*          vtmp = (float*)(ws + alloc((size_t)BATCH * 10 * 16 * 4));

    prep_w1<<<(256 * K1) / 256, 256, 0, stream>>>(w1, w1n);
    prep_w2<<<(256 * K2) / 256, 256, 0, stream>>>(w2, w2n);

    conv1_wmma<<<dim3(N1 / PXT1, 256 / OCT1), 256, 0, stream>>>(w1n, x, b1, h1n);
    conv2_wmma<<<dim3(N2 / PXT2, 256 / OCT2), 256, 0, stream>>>(w2n, h1n, b2, s2);

    squash_primary<<<(BATCH * NCAPS) / 256, 256, 0, stream>>>(s2, u);

    const int nblog = BATCH * NCAPS * 10;
    zero_f32<<<(nblog + 255) / 256, 256, 0, stream>>>(blog, nblog);

    for (int it = 0; it < 3; ++it) {
        float* vdst = (it == 2) ? out : vtmp;
        routing_sv<<<BATCH, 256, 0, stream>>>(u, W, blog, vdst);
        if (it < 2)
            routing_blog<<<(nblog + 255) / 256, 256, 0, stream>>>(u, W, vtmp, blog);
    }
}
